// TraceUnitTokenizer_17317308137477
// MI455X (gfx1250) — matrix operand order repaired and checked by reading, not compiled, not run on hardware
//
#include <hip/hip_runtime.h>
#include <hip/hip_bf16.h>

typedef long long i64;
typedef __attribute__((ext_vector_type(16))) __bf16 bf16x16;
typedef __attribute__((ext_vector_type(8)))  float  floatx8;

#if __has_builtin(__builtin_amdgcn_s_wait_asynccnt)
#define WAIT_ASYNC() __builtin_amdgcn_s_wait_asynccnt(0)
#else
#define WAIT_ASYNC() asm volatile("s_wait_asynccnt 0x0" ::: "memory")
#endif

union Frag {
  bf16x16 v;
  unsigned int u[8];
};

__device__ __forceinline__ unsigned short f2bf(float x) {
  unsigned u = __builtin_bit_cast(unsigned, x);
  u += 0x7FFFu + ((u >> 16) & 1u);     // round-to-nearest-even
  return (unsigned short)(u >> 16);
}

// pack 8 contiguous f32 -> 4 dwords of bf16 pairs
__device__ __forceinline__ void pack8(Frag& f, int off, const float* p) {
#pragma unroll
  for (int i = 0; i < 4; ++i)
    f.u[off + i] = (unsigned)f2bf(p[2 * i]) | ((unsigned)f2bf(p[2 * i + 1]) << 16);
}

// async 16-byte global -> LDS copy (ASYNCcnt); dst must be in LDS.
__device__ __forceinline__ void async_copy_b128(const void* gsrc, void* lds_dst) {
  unsigned lds_off = (unsigned)(__SIZE_TYPE__)lds_dst;  // low 32b of flat = LDS addr
  asm volatile("global_load_async_to_lds_b128 %0, %1, off"
               :
               : "v"(lds_off), "v"(gsrc)
               : "memory");
}

// =====================================================================
// prep: W_in (f32 [1420][384]) -> WiT bf16 [384][1440]  (transposed+padded)
// =====================================================================
__global__ __launch_bounds__(256) void prep_wit(const float* __restrict__ Wi,
                                                unsigned short* __restrict__ WiT) {
  int i = blockIdx.x * 256 + threadIdx.x;
  if (i >= 384 * 1440) return;
  int c = i / 1440, f = i - c * 1440;
  WiT[i] = f2bf((f < 1420) ? Wi[(i64)f * 384 + c] : 0.0f);
}

// =====================================================================
// Kernel A: token_features = concat(...) @ W_in + b_in    [65536 x 384]
// WG = 256 thr (8 waves), 64 token rows x 384 cols, double-buffered LDS.
// =====================================================================
__global__ __launch_bounds__(256) void feat_gemm(
    const float* __restrict__ bb, const float* __restrict__ st,
    const float* __restrict__ sem, const float* __restrict__ obj,
    const float* __restrict__ inst, const unsigned short* __restrict__ WiT,
    const float* __restrict__ bin, float* __restrict__ out_feat) {
  __shared__ __align__(16) unsigned short As[2][64 * 32];
  __shared__ __align__(16) unsigned short Bs[2][384 * 32];

  const int tid = threadIdx.x;
  const int lane = tid & 31;
  const int wave = tid >> 5;
  const int rg = wave >> 1;      // row group (16 rows each)
  const int cg = wave & 1;       // col group (192 cols each)
  const i64 wg_tok = (i64)blockIdx.x * 64;

  const floatx8 zero8 = {0.f, 0.f, 0.f, 0.f, 0.f, 0.f, 0.f, 0.f};
  floatx8 acc[12];
#pragma unroll
  for (int i = 0; i < 12; ++i) acc[i] = zero8;

  // ---- staging helpers ----
  const int srow = tid >> 2;             // A row this thread stages
  const int skk = (tid & 3) * 8;         // 8 consecutive k within the row
  auto stageA = [&](int kc, int buf) {
    i64 tok = wg_tok + srow;
    int b = (int)(tok >> 13);
    int rem = (int)(tok & 8191);
    int t = rem >> 7, k = rem & 127;
    unsigned tmp[4];
#pragma unroll
    for (int i = 0; i < 8; ++i) {
      int f = kc + skk + i;
      float val;
      if (f < 1152)        val = bb[tok * 1152 + f];
      else if (f < 1408)   val = sem[((i64)b * 128 + k) * 256 + (f - 1152)];
      else if (f < 1416)   val = st[tok * 8 + (f - 1408)];
      else if (f == 1416)  val = obj[b * 128 + k];
      else if (f == 1417)  val = inst[((i64)b * 128 + k) * 64 + t];
      else if (f == 1418)  val = (float)t * (1.0f / 63.0f);
      else if (f == 1419)  val = 1.0f - (float)t * (1.0f / 63.0f);
      else                 val = 0.0f;   // K padding 1420..1439
      unsigned bfv = f2bf(val);
      if ((i & 1) == 0) tmp[i >> 1] = bfv;
      else              tmp[i >> 1] |= bfv << 16;
    }
    *(uint4*)&As[buf][srow * 32 + skk] = *(const uint4*)tmp;
    // prefetch backbone stream two chunks ahead (global_prefetch_b8)
    if (kc + skk + 64 < 1152) __builtin_prefetch(&bb[tok * 1152 + kc + skk + 64], 0, 1);
  };
  auto stageB = [&](int kc, int buf) {
#pragma unroll
    for (int i = 0; i < 3; ++i) {
      int e = i * 256 + tid;             // 0..767 uint4s
      int col = e >> 2, kk = (e & 3) * 8;
      const unsigned short* src = WiT + (i64)col * 1440 + kc + kk;
      unsigned short* dst = &Bs[buf][col * 32 + kk];
      async_copy_b128(src, dst);         // ASYNCcnt-tracked LDS fill
    }
  };

  stageA(0, 0);
  stageB(0, 0);
  WAIT_ASYNC();
  __syncthreads();

  const int arow0 = rg * 16 + (lane & 15);
  const int kg = (lane < 16) ? 0 : 8;
  const int kb = (lane < 16) ? 0 : 16;

  for (int ic = 0; ic < 45; ++ic) {
    const int cur = ic & 1;
    // stage next chunk while computing current
    if (ic + 1 < 45) {
      stageA((ic + 1) * 32, cur ^ 1);
      stageB((ic + 1) * 32, cur ^ 1);
    }
    // A fragment (16x32 bf16, contiguous k = 16*half + s)
    Frag a;
    *(uint4*)&a.u[0] = *(const uint4*)&As[cur][arow0 * 32 + kb];
    *(uint4*)&a.u[4] = *(const uint4*)&As[cur][arow0 * 32 + kb + 8];
    // B fragments, software-pipelined
    Frag bf[2];
    {
      int col = cg * 192 + (lane & 15);
      *(uint4*)&bf[0].u[0] = *(const uint4*)&Bs[cur][col * 32 + kb];
      *(uint4*)&bf[0].u[4] = *(const uint4*)&Bs[cur][col * 32 + kb + 8];
    }
#pragma unroll
    for (int tI = 0; tI < 12; ++tI) {
      if (tI + 1 < 12) {
        int col = cg * 192 + (tI + 1) * 16 + (lane & 15);
        *(uint4*)&bf[(tI + 1) & 1].u[0] = *(const uint4*)&Bs[cur][col * 32 + kb];
        *(uint4*)&bf[(tI + 1) & 1].u[4] = *(const uint4*)&Bs[cur][col * 32 + kb + 8];
      }
      acc[tI] = __builtin_amdgcn_wmma_f32_16x16x32_bf16(
          false, a.v, false, bf[tI & 1].v, (short)0, acc[tI], false, false);
    }
    WAIT_ASYNC();
    __syncthreads();
  }

  // ---- epilogue: +b_in, store f32 (C layout: M = v + 8*hi, N = lane&15) ----
  const int hi = (lane < 16) ? 0 : 8;
#pragma unroll
  for (int tI = 0; tI < 12; ++tI) {
    int col = cg * 192 + tI * 16 + (lane & 15);
    float bv = bin[col];
#pragma unroll
    for (int v = 0; v < 8; ++v) {
      i64 tok = wg_tok + rg * 16 + v + hi;
      out_feat[tok * 384 + col] = acc[tI][v] + bv;
    }
  }
}

// =====================================================================
// units init: broadcast unit_queries to ws (f32 + bf16 mirror)
// =====================================================================
__global__ __launch_bounds__(256) void units_init(const float* __restrict__ uq,
                                                  float* __restrict__ ws_units,
                                                  unsigned short* __restrict__ ws_units_bf) {
  int i = blockIdx.x * 256 + threadIdx.x;
  if (i < 8 * 16 * 384) {
    float v = uq[i % (16 * 384)];
    ws_units[i] = v;
    ws_units_bf[i] = f2bf(v);
  }
}

// =====================================================================
// Kernel B: one slot iteration for 128 tokens per WG.
//   phase1: logits^T[m][tok] via WMMA (A=units bf16 rows, B=feature rows)
//   phase2: top-2 masked softmax per token (VALU), write assignment
//   phase3: partial updates[m][d] via WMMA (A=assignment^T, B=features)
// =====================================================================
__global__ __launch_bounds__(256) void slot_iter(
    const float* __restrict__ feat, const unsigned short* __restrict__ units_bf,
    const int* __restrict__ token_mask, const float* __restrict__ obj,
    const float* __restrict__ inst, float* __restrict__ out_assign,
    float* __restrict__ upd_part, float* __restrict__ denom_part) {
  __shared__ float logit_s[16 * 128];                    // [m][tok]
  __shared__ __align__(16) unsigned short aT[16 * 128];  // [m][tok] bf16
  __shared__ float bias_s[128];
  __shared__ float valid_s[128];
  __shared__ float denom_s[16];

  const int tid = threadIdx.x;
  const int lane = tid & 31;
  const int wave = tid >> 5;
  const int b = blockIdx.x >> 6;           // 64 WGs per batch
  const int base = (blockIdx.x & 63) * 128;

  if (tid < 16) denom_s[tid] = 0.0f;
  if (tid < 128) {
    int lt = base + tid;
    int tt = lt >> 7, kk = lt & 127;
    float o = obj[b * 128 + kk];
    float iv = inst[((i64)b * 128 + kk) * 64 + tt];
    bias_s[tid] = 0.1f * (o + iv);
    valid_s[tid] = token_mask[b * 128 + kk] ? 1.0f : 0.0f;
  }
  __syncthreads();

  // ---------- phase 1: logits^T tile (16 m x 16 tokens) per wave ----------
  {
    floatx8 c = {0.f, 0.f, 0.f, 0.f, 0.f, 0.f, 0.f, 0.f};
    const int m = lane & 15;
    const int kg = (lane < 16) ? 0 : 8;
    const int kb = (lane < 16) ? 0 : 16;
    const unsigned short* up = units_bf + ((i64)b * 16 + m) * 384;
    const float* fp = feat + ((i64)b * 8192 + base + wave * 16 + (lane & 15)) * 384;
#pragma unroll 4
    for (int kc = 0; kc < 384; kc += 32) {
      Frag a, bf;
      *(uint4*)&a.u[0] = *(const uint4*)&up[kc + kb];       // b128 from L2
      *(uint4*)&a.u[4] = *(const uint4*)&up[kc + kb + 8];
      pack8(bf, 0, fp + kc + kb);
      pack8(bf, 4, fp + kc + kb + 8);
      c = __builtin_amdgcn_wmma_f32_16x16x32_bf16(false, a.v, false, bf.v,
                                                  (short)0, c, false, false);
    }
    const float scale = 0.05103103630798288f;  // 1/sqrt(384)
    const float bv = bias_s[wave * 16 + (lane & 15)];
    const int hi = (lane < 16) ? 0 : 8;
#pragma unroll
    for (int v = 0; v < 8; ++v)
      logit_s[(v + hi) * 128 + wave * 16 + (lane & 15)] = c[v] * scale + bv;
  }
  __syncthreads();

  // ---------- phase 2: per-token top-2 + masked softmax ----------
  if (tid < 128) {
    float best = -3.4e38f, sec = -3.4e38f;
    int bi = 0, si = 1;
#pragma unroll
    for (int m = 0; m < 16; ++m) {
      float l = logit_s[m * 128 + tid];
      if (l > best) { sec = best; si = bi; best = l; bi = m; }
      else if (l > sec) { sec = l; si = m; }
    }
    float e2 = __expf((sec - best) * (1.0f / 0.7f));
    float inv = 1.0f / (1.0f + e2);
    float vm = valid_s[tid];
    float p1 = inv * vm;
    float p2 = e2 * inv * vm;
    i64 gtok = (i64)b * 8192 + base + tid;
#pragma unroll
    for (int m = 0; m < 16; ++m) {
      float a = (m == bi) ? p1 : ((m == si) ? p2 : 0.0f);
      out_assign[gtok * 16 + m] = a;
      aT[m * 128 + tid] = f2bf(a);
    }
    atomicAdd(&denom_s[bi], p1);
    atomicAdd(&denom_s[si], p2);
  }
  __syncthreads();
  if (tid < 16) atomicAdd(&denom_part[b * 16 + tid], denom_s[tid]);

  // ---------- phase 3: partial updates[m][d] (K = 128 tokens) ----------
  const int kg = (lane < 16) ? 0 : 8;
  const int kb = (lane < 16) ? 0 : 16;
  const int hi = (lane < 16) ? 0 : 8;
#pragma unroll
  for (int dts = 0; dts < 3; ++dts) {
    const int dt = wave * 3 + dts;              // 24 d-tiles over 8 waves
    const int d = dt * 16 + (lane & 15);
    floatx8 c = {0.f, 0.f, 0.f, 0.f, 0.f, 0.f, 0.f, 0.f};
#pragma unroll
    for (int t4 = 0; t4 < 4; ++t4) {
      Frag a, bf;
      *(uint4*)&a.u[0] = *(const uint4*)&aT[(lane & 15) * 128 + t4 * 32 + kb];
      *(uint4*)&a.u[4] = *(const uint4*)&aT[(lane & 15) * 128 + t4 * 32 + kb + 8];
#pragma unroll
      for (int v = 0; v < 8; ++v) {
        int tk = t4 * 32 + kb + 2 * v;
        float x0 = feat[((i64)b * 8192 + base + tk) * 384 + d];
        float x1 = feat[((i64)b * 8192 + base + tk + 1) * 384 + d];
        bf.u[v] = (unsigned)f2bf(x0) | ((unsigned)f2bf(x1) << 16);
      }
      c = __builtin_amdgcn_wmma_f32_16x16x32_bf16(false, a.v, false, bf.v,
                                                  (short)0, c, false, false);
    }
#pragma unroll
    for (int v = 0; v < 8; ++v)
      atomicAdd(&upd_part[((i64)b * 16 + v + hi) * 384 + d], c[v]);
  }
}

// =====================================================================
// Kernel C: units <- LN(units + MLP(LN(updates/denom)))   (128 rows)
// =====================================================================
__device__ __forceinline__ float block_reduce(float v, float* red) {
  int tid = threadIdx.x;
  red[tid] = v;
  __syncthreads();
#pragma unroll
  for (int s = 128; s > 0; s >>= 1) {
    if (tid < s) red[tid] += red[tid + s];
    __syncthreads();
  }
  float r = red[0];
  __syncthreads();
  return r;
}

__global__ __launch_bounds__(256) void units_update(
    float* __restrict__ ws_units, unsigned short* __restrict__ ws_units_bf,
    const float* __restrict__ upd_part, const float* __restrict__ denom_part,
    const float* __restrict__ uln_g, const float* __restrict__ uln_b,
    const float* __restrict__ W1, const float* __restrict__ b1,
    const float* __restrict__ W2, const float* __restrict__ b2,
    const float* __restrict__ nln_g, const float* __restrict__ nln_b,
    float* __restrict__ out_units, int write_out) {
  __shared__ float hbuf[384];
  __shared__ float mid[768];
  __shared__ float xbuf[384];
  __shared__ float red[256];

  const int r = blockIdx.x;  // b*16 + m
  const int tid = threadIdx.x;
  const float* urow = ws_units + (i64)r * 384;
  const float* prow = upd_part + (i64)r * 384;
  const float den = 1.0f / fmaxf(denom_part[r], 1e-6f);

  float s = 0.0f;
  for (int d = tid; d < 384; d += 256) { float v = prow[d] * den; hbuf[d] = v; s += v; }
  float mu = block_reduce(s, red) * (1.0f / 384.0f);
  float s2 = 0.0f;
  for (int d = tid; d < 384; d += 256) { float dv = hbuf[d] - mu; s2 += dv * dv; }
  float rstd = rsqrtf(block_reduce(s2, red) * (1.0f / 384.0f) + 1e-5f);
  __syncthreads();
  for (int d = tid; d < 384; d += 256)
    hbuf[d] = (hbuf[d] - mu) * rstd * uln_g[d] + uln_b[d];
  __syncthreads();

  for (int j = tid; j < 768; j += 256) {
    float acc = b1[j];
    for (int d = 0; d < 384; ++d) acc += hbuf[d] * W1[(i64)d * 768 + j];
    mid[j] = 0.5f * acc * (1.0f + erff(acc * 0.7071067811865475f));  // exact GELU
  }
  __syncthreads();

  for (int d = tid; d < 384; d += 256) {
    float acc = b2[d];
    for (int j = 0; j < 768; ++j) acc += mid[j] * W2[(i64)j * 384 + d];
    xbuf[d] = urow[d] + acc;
  }
  __syncthreads();

  float t = 0.0f;
  for (int d = tid; d < 384; d += 256) t += xbuf[d];
  float mu2 = block_reduce(t, red) * (1.0f / 384.0f);
  float t2 = 0.0f;
  for (int d = tid; d < 384; d += 256) { float dv = xbuf[d] - mu2; t2 += dv * dv; }
  float rstd2 = rsqrtf(block_reduce(t2, red) * (1.0f / 384.0f) + 1e-5f);
  __syncthreads();
  for (int d = tid; d < 384; d += 256) {
    float un = (xbuf[d] - mu2) * rstd2 * nln_g[d] + nln_b[d];
    ws_units[(i64)r * 384 + d] = un;
    ws_units_bf[(i64)r * 384 + d] = f2bf(un);
    if (write_out) out_units[(i64)r * 384 + d] = un;
  }
}

// =====================================================================
extern "C" void kernel_launch(void* const* d_in, const int* in_sizes, int n_in,
                              void* d_out, int out_size, void* d_ws, size_t ws_size,
                              hipStream_t stream) {
  (void)in_sizes; (void)n_in; (void)out_size; (void)ws_size;
  const float* bb    = (const float*)d_in[0];
  const float* st    = (const float*)d_in[1];
  const float* sem   = (const float*)d_in[2];
  const int*   tmask = (const int*)d_in[3];
  const float* obj   = (const float*)d_in[4];
  const float* inst  = (const float*)d_in[5];
  const float* Wi    = (const float*)d_in[6];
  const float* bin   = (const float*)d_in[7];
  const float* uq    = (const float*)d_in[8];
  const float* uln_g = (const float*)d_in[9];
  const float* uln_b = (const float*)d_in[10];
  const float* W1    = (const float*)d_in[11];
  const float* b1    = (const float*)d_in[12];
  const float* W2    = (const float*)d_in[13];
  const float* b2    = (const float*)d_in[14];
  const float* nln_g = (const float*)d_in[15];
  const float* nln_b = (const float*)d_in[16];

  float* out_feat   = (float*)d_out;                      // 65536*384
  float* out_assign = out_feat + (i64)65536 * 384;        // 65536*16
  float* out_units  = out_assign + (i64)65536 * 16;       // 128*384

  float* ws_units = (float*)d_ws;                              // 49152 f32
  float* ws_upd   = ws_units + 49152;                          // 49152 f32
  float* ws_den   = ws_upd + 49152;                            // 128 f32
  unsigned short* ws_units_bf = (unsigned short*)(ws_den + 128);   // 49152 bf16
  unsigned short* ws_WiT      = ws_units_bf + 49152;               // 384*1440 bf16

  prep_wit<<<2160, 256, 0, stream>>>(Wi, ws_WiT);
  feat_gemm<<<1024, 256, 0, stream>>>(bb, st, sem, obj, inst, ws_WiT, bin, out_feat);
  units_init<<<192, 256, 0, stream>>>(uq, ws_units, ws_units_bf);

  for (int it = 0; it < 3; ++it) {
    (void)hipMemsetAsync(ws_upd, 0, (49152 + 128) * sizeof(float), stream);
    slot_iter<<<512, 256, 0, stream>>>(out_feat, ws_units_bf, tmask, obj, inst,
                                       out_assign, ws_upd, ws_den);
    units_update<<<128, 256, 0, stream>>>(ws_units, ws_units_bf, ws_upd, ws_den,
                                          uln_g, uln_b, W1, b1, W2, b2, nln_g,
                                          nln_b, out_units, (it == 2) ? 1 : 0);
  }
}
